// GATLayer_40785009442965
// MI455X (gfx1250) — matrix operand order repaired and checked by reading, not compiled, not run on hardware
//
#include <hip/hip_runtime.h>
#include <hip/hip_bf16.h>

typedef __attribute__((ext_vector_type(16))) _Float16     v16h;
typedef __attribute__((ext_vector_type(8)))  float        v8f;
typedef __attribute__((ext_vector_type(4)))  unsigned int u32x4;
typedef __attribute__((ext_vector_type(8)))  int          i32x8;
typedef __attribute__((ext_vector_type(4)))  int          i32x4;

#define Bb   4
#define Nn   2048
#define IND  256
#define Hh   8
#define Dd   64
#define HD   512   // H*D
#define BN   8192  // B*N

__device__ __forceinline__ float leaky(float x) {
    return (x >= 0.f) ? x : 0.2f * x;
}

// ---------------------------------------------------------------------------
// Kernel 1: mapped = nodes @ proj_weights, stored transposed as f16:
//   mapped_t[(bh*64 + d)*2048 + node]
// One wave (32 threads) computes one 16x16 output tile via WMMA f16, K=256.
// ---------------------------------------------------------------------------
__global__ __launch_bounds__(32)
void proj_kernel(const float* __restrict__ nodes,
                 const float* __restrict__ pw,
                 _Float16*    __restrict__ mapped_t) {
    const int tile_m = blockIdx.x;        // 0..511  (rows of 16 over B*N)
    const int tile_n = blockIdx.y;        // 0..31   (cols of 16 over 512)
    const int lane   = threadIdx.x;
    const int m      = lane & 15;
    const int half   = lane >> 4;
    const int row    = tile_m * 16 + m;   // global row in [0, 8192)
    const int col0   = tile_n * 16;

    v8f acc = {};
    for (int kb = 0; kb < IND; kb += 32) {
        // --- A tile (16x32 f16): lane holds row m; K per ISA layout ---
        const float* arow = nodes + (size_t)row * IND + kb;
        v16h A;
#pragma unroll
        for (int e = 0; e < 8; ++e) {
            A[e]     = (_Float16)arow[half * 8 + e];        // K = half*8 + e
            A[8 + e] = (_Float16)arow[16 + half * 8 + e];   // K = 16 + half*8 + e
        }
        // --- B tile (32x16 f16): lane n = m, K-half = half ---
        v16h Bv;
#pragma unroll
        for (int kk = 0; kk < 16; ++kk) {
            Bv[kk] = (_Float16)pw[(size_t)(kb + ((kk < 8) ? (half * 8 + kk) : (16 + half * 8 + (kk - 8)))) * HD + col0 + m];
        }
        acc = __builtin_amdgcn_wmma_f32_16x16x32_f16(
                  false, A, false, Bv, (short)0, acc, false, false);
    }

    // --- store C tile to transposed f16 layout ---
#pragma unroll
    for (int r = 0; r < 8; ++r) {
        const int rr   = r + 8 * half;               // row within tile
        const int grow = tile_m * 16 + rr;           // b*2048 + node
        const int b    = grow >> 11;
        const int node = grow & (Nn - 1);
        const int c    = col0 + m;                   // 0..511
        const int h    = c >> 6;
        const int d    = c & 63;
        const size_t idx = (((size_t)(b * Hh + h)) * Dd + d) * Nn + node;
        mapped_t[idx] = (_Float16)acc[r];
    }
}

// ---------------------------------------------------------------------------
// Kernel 1b: el/er per (b,h,node) = dot(mapped[node,:], a_l / a_r)
// ---------------------------------------------------------------------------
__global__ __launch_bounds__(256)
void elr_kernel(const _Float16* __restrict__ mapped_t,
                const float*    __restrict__ aw,   // (H, 2D)
                float* __restrict__ el,
                float* __restrict__ er) {
    const int t    = blockIdx.x * 256 + threadIdx.x;   // 0 .. B*H*N-1
    const int bh   = t >> 11;
    const int node = t & (Nn - 1);
    const int h    = bh & (Hh - 1);
    const _Float16* base = mapped_t + (size_t)bh * Dd * Nn + node;
    float sl = 0.f, sr = 0.f;
#pragma unroll 8
    for (int d = 0; d < Dd; ++d) {
        const float v = (float)base[(size_t)d * Nn];
        sl += v * aw[h * (2 * Dd) + d];
        sr += v * aw[h * (2 * Dd) + Dd + d];
    }
    el[t] = sl;
    er[t] = sr;
}

// ---------------------------------------------------------------------------
// Kernel 2: masked softmax(leaky(el_i + er_j)) @ mapped, then sigmoid.
// Grid: (N/64, B*H). Block = 128 threads (4 waves); each wave owns 16 rows.
// LDS: full mapped_t[b,h] slab (64x2048 f16 = 256KB) + er (8KB) + reduce pad.
// The slab is DMA'd in with the Tensor Data Mover (tensor_load_to_lds).
// Softmax max = leaky(el_i + max_j er_j) (valid upper bound) -> no rescaling.
// ---------------------------------------------------------------------------
__global__ __launch_bounds__(128)
void attn_kernel(const int*      __restrict__ edges,
                 const _Float16* __restrict__ mapped_t,
                 const float*    __restrict__ el,
                 const float*    __restrict__ er,
                 float*          __restrict__ out) {
    extern __shared__ char smem[];
    _Float16* lmap = (_Float16*)smem;                       // 64*2048 f16
    float*    ler  = (float*)(smem + (size_t)Dd * Nn * 2);  // 2048 f32
    float*    lred = ler + Nn;                              // 4 f32

    const int bh = blockIdx.y;
    const int b  = bh >> 3;
    const int h  = bh & (Hh - 1);
    const int i0 = blockIdx.x * 64;
    const int tid = threadIdx.x;

    // ---- preload mapped slab via TDM (one DMA issued by wave 0) ----
    const unsigned long long gaddr =
        (unsigned long long)(const void*)(mapped_t + (size_t)bh * Dd * Nn);
    if (tid < 32) {
        // D# group 0: count=1 | lds_addr | global_addr[56:0] | type=2
        const unsigned ldsaddr = (unsigned)(unsigned long long)(void*)lmap;
        u32x4 g0;
        g0[0] = 1u;                                     // count=1, user mode
        g0[1] = ldsaddr;                                // lds_addr (bytes)
        g0[2] = (unsigned)(gaddr & 0xFFFFFFFFull);      // global_addr[31:0]
        g0[3] = (unsigned)((gaddr >> 32) & 0x01FFFFFFull) | (2u << 30); // type=2
        // D# group 1: data_size=2B, 2D tensor 2048 x 64, tile 2048 x 64
        i32x8 g1;
        g1[0] = 0x10000;                 // data_size=1 (2 bytes); no mask/flags
        g1[1] = (int)((Nn & 0xFFFF) << 16);             // tensor_dim0 lo16
        g1[2] = (int)((Nn >> 16) | ((Dd & 0xFFFF) << 16)); // dim0 hi / dim1 lo
        g1[3] = (int)((Dd >> 16) | ((Nn & 0xFFFF) << 16)); // dim1 hi / tile_dim0
        g1[4] = (int)(Dd & 0xFFFF);                     // tile_dim1=64, tile_dim2=0
        g1[5] = (int)Nn;                                // dim0_stride lo32 = 2048
        g1[6] = (int)(((Dd * Nn) & 0xFFFF) << 16);      // stride0 hi / stride1 lo16
        g1[7] = (int)((Dd * Nn) >> 16);                 // stride1 hi
        i32x4 g2 = {0, 0, 0, 0};
        i32x4 g3 = {0, 0, 0, 0};
        i32x8 g4 = {0, 0, 0, 0, 0, 0, 0, 0};
        __builtin_amdgcn_tensor_load_to_lds(g0, g1, g2, g3, g4, 0);
        __builtin_amdgcn_s_wait_tensorcnt(0);
    }
    // ---- er preload (all threads) ----
    {
        const float* erb = er + bh * Nn;
        for (int i = tid; i < Nn; i += 128) ler[i] = erb[i];
    }
    __syncthreads();

    // ---- er max reduction (for the safe softmax shift) ----
    float mx = -1e30f;
    for (int i = tid; i < Nn; i += 128) mx = fmaxf(mx, ler[i]);
#pragma unroll
    for (int off = 16; off > 0; off >>= 1) mx = fmaxf(mx, __shfl_xor(mx, off, 32));
    if ((tid & 31) == 0) lred[tid >> 5] = mx;
    __syncthreads();
    const float ermax = fmaxf(fmaxf(lred[0], lred[1]), fmaxf(lred[2], lred[3]));

    // ---- per-wave state: 16 rows each ----
    const int wave = tid >> 5;
    const int lane = tid & 31;
    const int m    = lane & 15;
    const int half = lane >> 4;
    const int irow = i0 + wave * 16 + m;                 // node row for this lane
    const float elv  = el[bh * Nn + irow];
    const float mrow = leaky(elv + ermax);               // softmax shift (>= all scores)

    v8f acc[4] = {{}, {}, {}, {}};
    float psum = 0.f;
    const int* erow = edges + ((size_t)b * Nn + irow) * Nn;

    for (int jb = 0; jb < Nn; jb += 32) {
        // prefetch the edge chunk ~4 iterations ahead (wrapped: always in-bounds)
        __builtin_prefetch(erow + ((jb + 128) & (Nn - 1)), 0, 1);

        // ---- build probability A-tile (16x32 f16), K = 16*half + e (contiguous) ----
        v16h A;
        float p[16];
#pragma unroll
        for (int e = 0; e < 16; ++e) {
            const int j = jb + half * 16 + e;
            float s = elv + ler[j];
            s = leaky(s);
            float pe = __expf(s - mrow);
            pe = (erow[j] != 0) ? pe : 0.f;
            p[e] = pe;
            psum += pe;
        }
#pragma unroll
        for (int e = 0; e < 16; ++e) A[e] = (_Float16)p[e];

        // ---- 4 B-tiles (32x16 f16) straight out of LDS (contiguous) ----
#pragma unroll
        for (int t = 0; t < 4; ++t) {
            const int d = t * 16 + m;                          // column = lane%16
            const v16h Bv = *(const v16h*)(lmap + (size_t)d * Nn + jb + half * 16);
            acc[t] = __builtin_amdgcn_wmma_f32_16x16x32_f16(
                         false, A, false, Bv, (short)0, acc[t], false, false);
        }
    }

    // ---- combine row sums across the two K-halves ----
    float rowtot = psum + __shfl_xor(psum, 16, 32);   // lane L holds total of row L%16

    // ---- normalize, sigmoid, store (C layout: VGPR r -> row r+8*half, col m) ----
#pragma unroll
    for (int r = 0; r < 8; ++r) {
        const int rr = r + 8 * half;
        float denom = __shfl(rowtot, rr, 32);
        denom = fmaxf(denom, 1e-30f);
        const int orow = i0 + wave * 16 + rr;
#pragma unroll
        for (int t = 0; t < 4; ++t) {
            const float v  = acc[t][r] / denom;
            const float sg = 1.f / (1.f + __expf(-v));
            out[((size_t)b * Nn + orow) * HD + h * Dd + t * 16 + m] = sg;
        }
    }
}

// ---------------------------------------------------------------------------
extern "C" void kernel_launch(void* const* d_in, const int* in_sizes, int n_in,
                              void* d_out, int out_size, void* d_ws, size_t ws_size,
                              hipStream_t stream) {
    const float* nodes = (const float*)d_in[0];   // (4,2048,256) f32
    const int*   edges = (const int*)  d_in[1];   // (4,2048,2048) i32
    const float* pw    = (const float*)d_in[2];   // (256,512) f32
    const float* aw    = (const float*)d_in[3];   // (8,128) f32
    float* out = (float*)d_out;                   // (4,2048,512) f32

    // workspace layout
    _Float16* mapped_t = (_Float16*)d_ws;                          // 8 MB
    float* el = (float*)((char*)d_ws + (size_t)Bb * Hh * Dd * Nn * 2);
    float* er = el + Bb * Hh * Nn;

    // 1) projection GEMM -> mapped_t (f16, transposed)
    {
        dim3 grid(BN / 16, HD / 16);   // 512 x 32
        proj_kernel<<<grid, 32, 0, stream>>>(nodes, pw, mapped_t);
    }
    // 1b) el / er
    {
        dim3 grid((Bb * Hh * Nn) / 256);   // 256 blocks
        elr_kernel<<<grid, 256, 0, stream>>>(mapped_t, aw, el, er);
    }
    // 2) attention + softmax + output
    {
        dim3 grid(Nn / 64, Bb * Hh);       // 32 x 32
        const size_t shmem = (size_t)Dd * Nn * 2   // mapped slab (f16)
                           + (size_t)Nn * 4        // er
                           + 64;                   // reduce pad
        attn_kernel<<<grid, 128, shmem, stream>>>(edges, mapped_t, el, er, out);
    }
}
